// Gauss_10831907520642
// MI455X (gfx1250) — matrix operand order repaired and checked by reading, not compiled, not run on hardware
//
#include <hip/hip_runtime.h>
#include <hip/hip_bf16.h>
#include <math.h>

// GMM diagonal-covariance log-likelihood as one fused bf16 WMMA GEMM.
//   out[m,k] = bias[k] + sum_f x^2*(-0.5/cov) + sum_f x*(mu/cov)
// A(16384 x 2048 bf16) = [x^2 | x], B(512 x 2048 bf16) = [-0.5/cov | mu/cov].
// GEMM feeds v_wmma_f32_16x16x32_bf16 from LDS tiles staged by the Tensor
// Data Mover (tensor_load_to_lds, TENSORcnt double-buffered).

#define GMM_F 1024
#define GMM_K 500
#define GMM_KPAD 512
#define GMM_M 16384
#define GMM_KDIM 2048

#define TILE_K 64                   // K elements staged per TDM stage
#define NSTAGE (GMM_KDIM / TILE_K)  // 32
#define LDS_ROW 144                 // 128B row + 16B TDM pad -> 36-bank stride
#define LDS_TILE (128 * LDS_ROW)    // 18432 B per tile
#define AOFF(p) ((unsigned)((p) * LDS_TILE))
#define BOFF(p) ((unsigned)(2 * LDS_TILE + (p) * LDS_TILE))
#define SMEM_BYTES (4 * LDS_TILE)   // 73728 B (of 320KB/WGP)

typedef __attribute__((ext_vector_type(16))) __bf16 v16bf;
typedef __attribute__((ext_vector_type(8)))  __bf16 v8bf;
typedef __attribute__((ext_vector_type(4)))  __bf16 v4bf;
typedef __attribute__((ext_vector_type(8)))  float  v8f;

#if defined(__has_builtin)
#if __has_builtin(__builtin_amdgcn_tensor_load_to_lds) && \
    __has_builtin(__builtin_amdgcn_s_wait_tensorcnt)
#define GMM_USE_TDM 1
#endif
#endif
#ifndef GMM_USE_TDM
#define GMM_USE_TDM 0
#endif

static __device__ __forceinline__ v8f wmma_bf16(v16bf a, v16bf b, v8f c) {
  return __builtin_amdgcn_wmma_f32_16x16x32_bf16(false, a, false, b, (short)0, c,
                                                 false, false);
}

#if GMM_USE_TDM
typedef __attribute__((ext_vector_type(4))) unsigned int u32x4;
typedef __attribute__((ext_vector_type(8))) int i32x8;
typedef __attribute__((ext_vector_type(4))) int i32x4;

// Issue one TDM 2-D tile load: tile_dim0=64 elems (128B) x tile_dim1=128 rows,
// bf16 tensor row length GMM_KDIM, LDS pad 16B per 128B row (D# per ISA ch.8).
static __device__ __forceinline__ void tdm_load_tile(unsigned lds_addr,
                                                     const __bf16* gsrc,
                                                     unsigned tensor_rows) {
  const unsigned long long ga = (unsigned long long)(uintptr_t)gsrc;
  u32x4 g0;
  g0[0] = 1u;                                   // count=1, user descriptor
  g0[1] = lds_addr;                             // LDS byte address
  g0[2] = (unsigned)ga;                         // global_addr[31:0]
  g0[3] = (unsigned)((ga >> 32) & 0x1FFFFFFu)   // global_addr[56:32]
          | (2u << 30);                         // type = 2 ("image")
  i32x8 g1;
  g1[0] = (int)((1u << 16)      // data_size = 1 -> 2 bytes
                | (1u << 20)    // pad_enable
                | (4u << 22)    // pad_interval: 2^4 * 8B = 128B
                | (3u << 25));  // pad_amount: 4 DWORDs = 16B
  g1[1] = (int)(((unsigned)GMM_KDIM & 0xFFFFu) << 16);        // tensor_dim0 lo
  g1[2] = (int)(((unsigned)GMM_KDIM >> 16) & 0xFFFFu          // tensor_dim0 hi
                | ((tensor_rows & 0xFFFFu) << 16));           // tensor_dim1 lo
  g1[3] = (int)(((tensor_rows >> 16) & 0xFFFFu)               // tensor_dim1 hi
                | ((unsigned)TILE_K << 16));                  // tile_dim0 = 64
  g1[4] = (int)128u;                                          // tile_dim1 = 128
  g1[5] = (int)(unsigned)GMM_KDIM;                            // dim0_stride lo32
  g1[6] = 0;                                                  // stride hi bits
  g1[7] = 0;
  const i32x4 z4 = {0, 0, 0, 0};
  const i32x8 z8 = {0, 0, 0, 0, 0, 0, 0, 0};
  // clang-23 / therock-10.0 6-arg form: (g0, g1, g2, g3, extra_g, cpol)
  __builtin_amdgcn_tensor_load_to_lds(g0, g1, z4, z4, z8, 0);
}
#endif

// ---------------------------------------------------------------------------
// Kernel 1: unified B matrix (bf16 [512][2048]) + per-cluster bias.
// ---------------------------------------------------------------------------
__global__ void gmm_prep_kernel(const float* __restrict__ mu,
                                const float* __restrict__ cov,
                                __bf16* __restrict__ Bm,
                                float* __restrict__ bias) {
  const int k   = blockIdx.x;
  const int tid = threadIdx.x;
  __shared__ float s_log[256];
  __shared__ float s_muq[256];

  __bf16* __restrict__ Brow = Bm + (size_t)k * GMM_KDIM;

  if (k >= GMM_K) {
    for (int f = tid; f < GMM_F; f += 256) {
      Brow[f]         = (__bf16)0.0f;
      Brow[GMM_F + f] = (__bf16)0.0f;
    }
    if (tid == 0) bias[k] = 0.0f;
    return;
  }

  float slog = 0.0f, smuq = 0.0f;
  for (int f = tid; f < GMM_F; f += 256) {
    const size_t idx = (size_t)k * GMM_F + f;
    const float c  = cov[idx];
    const float m  = mu[idx];
    const float ic = 1.0f / c;
    Brow[f]         = (__bf16)(-0.5f * ic);
    Brow[GMM_F + f] = (__bf16)(m * ic);
    slog += logf(c);
    smuq += m * m * ic;
  }
  s_log[tid] = slog;
  s_muq[tid] = smuq;
  __syncthreads();
  for (int off = 128; off > 0; off >>= 1) {
    if (tid < off) {
      s_log[tid] += s_log[tid + off];
      s_muq[tid] += s_muq[tid + off];
    }
    __syncthreads();
  }
  if (tid == 0) {
    const float LOG_2PI = 1.8378770664093453f;
    bias[k] = -0.5f * (float)GMM_F * LOG_2PI - 0.5f * s_log[0] - 0.5f * s_muq[0];
  }
}

// ---------------------------------------------------------------------------
// Kernel 2: pack A once: fp32 X row (1024) -> bf16 row (2048) = [x^2 | x].
// ---------------------------------------------------------------------------
__global__ void __launch_bounds__(256)
gmm_packA_kernel(const float* __restrict__ X, __bf16* __restrict__ A) {
  const int gid = blockIdx.x * 256 + threadIdx.x;
  const int row = gid >> 8;
  const int c4  = (gid & 255) * 4;
  const float4 v = *(const float4*)(X + (size_t)row * GMM_F + c4);
  v4bf q, l;
  q[0] = (__bf16)(v.x * v.x); q[1] = (__bf16)(v.y * v.y);
  q[2] = (__bf16)(v.z * v.z); q[3] = (__bf16)(v.w * v.w);
  l[0] = (__bf16)v.x; l[1] = (__bf16)v.y;
  l[2] = (__bf16)v.z; l[3] = (__bf16)v.w;
  __bf16* Arow = A + (size_t)row * GMM_KDIM;
  *(v4bf*)(Arow + c4)         = q;
  *(v4bf*)(Arow + GMM_F + c4) = l;
}

// ---------------------------------------------------------------------------
// Kernel 3: WMMA GEMM, TDM/LDS double-buffered.
// Grid (512/128, 16384/128); block 256 = 8 waves, 2(M) x 4(N); each wave owns
// 64x32 = 4x2 accumulators. Stage = 64 K-elements -> 2 WMMA K-steps of 32.
// ---------------------------------------------------------------------------
__global__ void __launch_bounds__(256)
gmm_gemm_kernel(const __bf16* __restrict__ A,   // [16384][2048]
                const __bf16* __restrict__ Bm,  // [512][2048]
                const float* __restrict__ bias,
                float* __restrict__ out) {
  __shared__ __align__(16) unsigned char smem[SMEM_BYTES];

  const int tid  = threadIdx.x;
  const int lane = tid & 31;
  const int wave = tid >> 5;
  const int wm   = wave & 1;   // M half (64 rows of the 128-row tile)
  const int wn   = wave >> 1;  // N quarter (32 cols of the 128-col tile)
  const int l15  = lane & 15;
  const int hi   = lane >> 4;

  const int rowTile = blockIdx.y * 128;
  const int colTile = blockIdx.x * 128;
  const __bf16* Ag = A  + (size_t)rowTile * GMM_KDIM;  // tile row 0, col 0
  const __bf16* Bg = Bm + (size_t)colTile * GMM_KDIM;

  v8f acc[4][2] = {{{}, {}}, {{}, {}}, {{}, {}}, {{}, {}}};

#if GMM_USE_TDM
  const bool issuer = (wave == 0);
  const unsigned smemBase = (unsigned)(uintptr_t)&smem[0];
  if (issuer) {
    tdm_load_tile(smemBase + AOFF(0), Ag, (unsigned)GMM_M);
    tdm_load_tile(smemBase + BOFF(0), Bg, (unsigned)GMM_KPAD);
  }
#endif

  for (int kt = 0; kt < NSTAGE; ++kt) {
#if GMM_USE_TDM
    const int p = kt & 1;
    if (issuer) {
      if (kt + 1 < NSTAGE) {
        tdm_load_tile(smemBase + AOFF(p ^ 1), Ag + (kt + 1) * TILE_K,
                      (unsigned)GMM_M);
        tdm_load_tile(smemBase + BOFF(p ^ 1), Bg + (kt + 1) * TILE_K,
                      (unsigned)GMM_KPAD);
        __builtin_amdgcn_s_wait_tensorcnt((short)2);  // stage kt complete
      } else {
        __builtin_amdgcn_s_wait_tensorcnt((short)0);
      }
    }
    __syncthreads();  // LDS stage p visible to all waves
#else
    const int p = 0;
    __syncthreads();
    {  // cooperative copy fallback: thread t -> row t/2, 32-elem half-row
      const int row = tid >> 1;
      const int ch  = (tid & 1) * 32;
      const __bf16* sa = Ag + (size_t)row * GMM_KDIM + kt * TILE_K + ch;
      const __bf16* sb = Bg + (size_t)row * GMM_KDIM + kt * TILE_K + ch;
      const unsigned da = AOFF(0) + (unsigned)row * LDS_ROW + ch * 2;
      const unsigned db = BOFF(0) + (unsigned)row * LDS_ROW + ch * 2;
#pragma unroll
      for (int j = 0; j < 4; ++j) {
        *(v8bf*)(smem + da + j * 16) = *(const v8bf*)(sa + j * 8);
        *(v8bf*)(smem + db + j * 16) = *(const v8bf*)(sb + j * 8);
      }
    }
    __syncthreads();
#endif

    // ---- compute: 2 WMMA K-steps of 32 from LDS stage p ----
#pragma unroll
    for (int ksub = 0; ksub < 2; ++ksub) {
      const unsigned kOffA = (unsigned)(ksub * 64 + (hi ? 32 : 0));  // bytes
      const unsigned kOffB = (unsigned)(ksub * 64 + (hi ? 32 : 0));  // bytes

      v16bf afr[4];
#pragma unroll
      for (int mt = 0; mt < 4; ++mt) {
        const unsigned base =
            AOFF(p) + (unsigned)(wm * 64 + mt * 16 + l15) * LDS_ROW + kOffA;
        const v8bf c0 = *(const v8bf*)(smem + base);
        const v8bf c1 = *(const v8bf*)(smem + base + 16);
        afr[mt] = __builtin_shufflevector(c0, c1, 0, 1, 2, 3, 4, 5, 6, 7, 8, 9,
                                          10, 11, 12, 13, 14, 15);
      }
      v16bf bfr[2];
#pragma unroll
      for (int nt = 0; nt < 2; ++nt) {
        const unsigned base =
            BOFF(p) + (unsigned)(wn * 32 + nt * 16 + l15) * LDS_ROW + kOffB;
        bfr[nt] = *(const v16bf*)(smem + base);
      }
#pragma unroll
      for (int mt = 0; mt < 4; ++mt)
#pragma unroll
        for (int nt = 0; nt < 2; ++nt)
          acc[mt][nt] = wmma_bf16(afr[mt], bfr[nt], acc[mt][nt]);
    }
    __syncthreads();  // all waves done with stage p before TDM overwrites it
  }

  // ---- epilogue: bias + guarded stores ----
  const int col0 = colTile + wn * 32 + l15;
  const int col1 = col0 + 16;
  const float bv0 = (col0 < GMM_K) ? bias[col0] : 0.0f;
  const float bv1 = (col1 < GMM_K) ? bias[col1] : 0.0f;

#pragma unroll
  for (int mt = 0; mt < 4; ++mt) {
#pragma unroll
    for (int r = 0; r < 8; ++r) {
      const int row = rowTile + wm * 64 + mt * 16 + r + 8 * hi;
      if (col0 < GMM_K) out[(size_t)row * GMM_K + col0] = acc[mt][0][r] + bv0;
      if (col1 < GMM_K) out[(size_t)row * GMM_K + col1] = acc[mt][1][r] + bv1;
    }
  }
}

extern "C" void kernel_launch(void* const* d_in, const int* in_sizes, int n_in,
                              void* d_out, int out_size, void* d_ws, size_t ws_size,
                              hipStream_t stream) {
  const float* X   = (const float*)d_in[0];  // (16,1024,1024)
  const float* mu  = (const float*)d_in[1];  // (500,1024)
  const float* cov = (const float*)d_in[2];  // (500,1024)
  float* out = (float*)d_out;                // (16,1024,500) fp32

  // Workspace: Apack (64MB) | Ball (2MB) | bias
  __bf16* Apack = (__bf16*)d_ws;
  __bf16* Ball  = Apack + (size_t)GMM_M * GMM_KDIM;
  float*  bias  = (float*)(Ball + (size_t)GMM_KPAD * GMM_KDIM);

  gmm_prep_kernel<<<dim3(GMM_KPAD), dim3(256), 0, stream>>>(mu, cov, Ball, bias);
  gmm_packA_kernel<<<dim3(GMM_M), dim3(256), 0, stream>>>(X, Apack);

  dim3 grid(GMM_KPAD / 128, GMM_M / 128);
  gmm_gemm_kernel<<<grid, dim3(256), 0, stream>>>(Apack, Ball, bias, out);
}
